// Policy_13675175871132
// MI455X (gfx1250) — matrix operand order repaired and checked by reading, not compiled, not run on hardware
//
#include <hip/hip_runtime.h>
#include <hip/hip_bf16.h>
#include <math.h>

#define T_STEPS 512
#define BATCH   256
#define OBS_D   64
#define HID     128
#define G4H     512      // 4*HID
#define KC      192      // OBS + HID combined contraction
#define NACT    4
#define NBINS   16
#define ALOG    64       // NACT*NBINS

typedef __attribute__((ext_vector_type(16))) _Float16 v16h;
typedef __attribute__((ext_vector_type(8)))  _Float16 v8h;
typedef __attribute__((ext_vector_type(8)))  float    v8f;

// ---- workspace layout (bytes) ----
#define WS_WC    0          // 512*192 f16   = 196608
#define WS_BIAS  196608     // 512 f32       = 2048
#define WS_AW1   198656     // 128*128 f16
#define WS_AW2   231424
#define WS_AW3   264192     // 64*128 f16
#define WS_CW1   280576
#define WS_CW2   313344
#define WS_HID   393216     // T*B*H f16 = 33554432

// A fragment: 16x32 f16, M = lane%16, K interleave: lanes<16 K={k0..k0+7, k0+16..k0+23}
__device__ __forceinline__ v16h frag_a(const _Float16* base, int stride, int row0, int k0, int lane) {
  const _Float16* p = base + (size_t)(row0 + (lane & 15)) * stride + k0 + ((lane & 16) ? 8 : 0);
  union { v16h v; v8h h[2]; } u;
  u.h[0] = *(const v8h*)(p);
  u.h[1] = *(const v8h*)(p + 16);
  return u.v;
}
// B fragment: 32x16 f16, N = lane%16, K interleave: lanes<16 K={k0..k0+7, k0+16..k0+23}
__device__ __forceinline__ v16h frag_b(const _Float16* base, int stride, int row0, int k0, int lane) {
  const _Float16* p = base + (size_t)(row0 + (lane & 15)) * stride + k0 + ((lane & 16) ? 8 : 0);
  union { v16h v; v8h h[2]; } u;
  u.h[0] = *(const v8h*)(p);
  u.h[1] = *(const v8h*)(p + 16);
  return u.v;
}

__device__ __forceinline__ float sigmoidf_(float x) { return 1.0f / (1.0f + __expf(-x)); }
__device__ __forceinline__ float tanhf_(float x)    { return 2.0f / (1.0f + __expf(-2.0f * x)) - 1.0f; }

// ---------------- weight prep: f32 -> f16, pack [w_ih|w_hh], fuse biases ----------------
__global__ void prep_kernel(const float* __restrict__ w_ih, const float* __restrict__ w_hh,
                            const float* __restrict__ b_ih, const float* __restrict__ b_hh,
                            const float* __restrict__ aw1, const float* __restrict__ aw2,
                            const float* __restrict__ aw3, const float* __restrict__ cw1,
                            const float* __restrict__ cw2, unsigned char* __restrict__ ws) {
  int i = blockIdx.x * blockDim.x + threadIdx.x;
  int j = i;
  if (j < G4H * KC) {                       // packed combined weights
    int r = j / KC, c = j % KC;
    float v = (c < OBS_D) ? w_ih[r * OBS_D + c] : w_hh[r * HID + (c - OBS_D)];
    ((_Float16*)(ws + WS_WC))[j] = (_Float16)v;
    return;
  }
  j -= G4H * KC;
  if (j < G4H) { ((float*)(ws + WS_BIAS))[j] = b_ih[j] + b_hh[j]; return; }
  j -= G4H;
  if (j < HID * HID) { ((_Float16*)(ws + WS_AW1))[j] = (_Float16)aw1[j]; return; }
  j -= HID * HID;
  if (j < HID * HID) { ((_Float16*)(ws + WS_AW2))[j] = (_Float16)aw2[j]; return; }
  j -= HID * HID;
  if (j < ALOG * HID) { ((_Float16*)(ws + WS_AW3))[j] = (_Float16)aw3[j]; return; }
  j -= ALOG * HID;
  if (j < HID * HID) { ((_Float16*)(ws + WS_CW1))[j] = (_Float16)cw1[j]; return; }
  j -= HID * HID;
  if (j < HID * HID) { ((_Float16*)(ws + WS_CW2))[j] = (_Float16)cw2[j]; return; }
}

// ---------------- masked LSTM scan ----------------
// 16 WGs x 16 batch rows. Wave w owns h-columns [16w,16w+16) across ALL four gates,
// so the i/f/g/o values for each (row,col) land in one lane's registers and the whole
// pointwise LSTM update is register-resident (c state never leaves VGPRs).
__global__ __launch_bounds__(256) void lstm_kernel(const float* __restrict__ obs,
                                                   const float* __restrict__ done,
                                                   const float* __restrict__ h0,
                                                   const float* __restrict__ c0,
                                                   const unsigned char* __restrict__ ws,
                                                   _Float16* __restrict__ hid) {
  alignas(16) __shared__ _Float16 rowbuf[16 * KC];   // [x(64) | h(128)] f16 per row
  __shared__ float mbuf[2][16];                      // double-buffered (1-done[t+1])
  const _Float16* wc    = (const _Float16*)(ws + WS_WC);
  const float*    biasg = (const float*)(ws + WS_BIAS);
  const int tid = threadIdx.x, lane = tid & 31, wave = tid >> 5;
  const int b0 = blockIdx.x * 16;
  const int cw = wave * 16 + (lane & 15);            // owned h column
  const int r0 = (lane & 16) ? 8 : 0;                // D-fragment row base

  // loop-invariant per-lane bias (folded into accumulator init)
  float bv[4];
#pragma unroll
  for (int g = 0; g < 4; ++g) bv[g] = biasg[g * HID + cw];

  // init: h0/c0 pre-masked by done[0] (reference masks at step start)
  float c_reg[8];
#pragma unroll
  for (int r = 0; r < 8; ++r) {
    int row = r0 + r;
    float m0 = 1.0f - done[b0 + row];
    c_reg[r] = c0[(b0 + row) * HID + cw] * m0;
  }
  for (int i = tid; i < 16 * HID; i += 256) {
    int r = i >> 7, n = i & 127;
    float m0 = 1.0f - done[b0 + r];
    rowbuf[r * KC + OBS_D + n] = (_Float16)(h0[(b0 + r) * HID + n] * m0);
  }

  for (int t = 0; t < T_STEPS; ++t) {
    // phase S: stage x_t (f16) + next-step mask; prefetch next obs rows
    for (int i = tid; i < 16 * OBS_D; i += 256) {
      int r = i >> 6, o = i & 63;
      rowbuf[r * KC + o] = (_Float16)obs[((size_t)t * BATCH + b0 + r) * OBS_D + o];
      if (t + 1 < T_STEPS)
        __builtin_prefetch(&obs[((size_t)(t + 1) * BATCH + b0 + r) * OBS_D + o], 0, 1);
    }
    if (tid < 16)
      mbuf[t & 1][tid] = (t + 1 < T_STEPS) ? (1.0f - done[(t + 1) * BATCH + b0 + tid]) : 1.0f;
    __syncthreads();

    // phase L: everyone snapshots the A fragments (full [x|h] rows)
    v16h afr[6];
#pragma unroll
    for (int kk = 0; kk < 6; ++kk) afr[kk] = frag_a(rowbuf, KC, 0, kk * 32, lane);
    __syncthreads();

    // phase C: four gate tiles for this wave's column slice; wc fragments are
    // loop-invariant -> compiler hoists them into high VGPRs.
    v8f acc[4];
#pragma unroll
    for (int g = 0; g < 4; ++g) {
#pragma unroll
      for (int r = 0; r < 8; ++r) acc[g][r] = bv[g];
    }
#pragma unroll
    for (int kk = 0; kk < 6; ++kk) {
#pragma unroll
      for (int g = 0; g < 4; ++g) {
        v16h bfr = frag_b(wc, KC, g * HID + wave * 16, kk * 32, lane);
        acc[g] = __builtin_amdgcn_wmma_f32_16x16x32_f16(false, afr[kk], false, bfr,
                                                        (short)0, acc[g], false, false);
      }
    }
    // register-resident pointwise update; fold done[t+1] into recurrent copies
#pragma unroll
    for (int r = 0; r < 8; ++r) {
      int row = r0 + r;
      float mnext = mbuf[t & 1][row];
      float iv = sigmoidf_(acc[0][r]);
      float fv = sigmoidf_(acc[1][r]);
      float gv = tanhf_(acc[2][r]);
      float ov = sigmoidf_(acc[3][r]);
      float cc = fv * c_reg[r] + iv * gv;
      float hh = ov * tanhf_(cc);
      c_reg[r] = cc * mnext;
      rowbuf[row * KC + OBS_D + cw] = (_Float16)(hh * mnext);         // masked recurrent h
      hid[((size_t)t * BATCH + b0 + row) * HID + cw] = (_Float16)hh;  // unmasked output
    }
  }
}

// ---------------- head GEMM helpers (M=64 per block, 8 waves) ----------------
__device__ __forceinline__ void gemm_tanh(const _Float16* inb, const _Float16* wg,
                                          const float* bias, int N, _Float16* outb,
                                          int lane, int wave) {
  const int ntiles = N >> 4;
  const int per = (4 * ntiles) >> 3;   // tiles per wave
  for (int j = 0; j < per; ++j) {
    int tt = wave * per + j;
    int mt = tt / ntiles, nt = tt % ntiles;
    int col = nt * 16 + (lane & 15);
    float bv = bias[col];
    v8f acc;
#pragma unroll
    for (int r = 0; r < 8; ++r) acc[r] = bv;
#pragma unroll
    for (int kk = 0; kk < 4; ++kk) {
      v16h a = frag_a(inb, HID, mt * 16, kk * 32, lane);
      v16h b = frag_b(wg, HID, nt * 16, kk * 32, lane);
      acc = __builtin_amdgcn_wmma_f32_16x16x32_f16(false, a, false, b, (short)0, acc, false, false);
    }
    int r0 = mt * 16 + ((lane & 16) ? 8 : 0);
#pragma unroll
    for (int r = 0; r < 8; ++r)
      outb[(r0 + r) * N + col] = (_Float16)tanhf_(acc[r]);
  }
}

__device__ __forceinline__ void gemm_lin(const _Float16* inb, const _Float16* wg,
                                         const float* bias, int N, float* outf,
                                         int lane, int wave) {
  const int ntiles = N >> 4;
  const int per = (4 * ntiles) >> 3;
  for (int j = 0; j < per; ++j) {
    int tt = wave * per + j;
    int mt = tt / ntiles, nt = tt % ntiles;
    int col = nt * 16 + (lane & 15);
    float bv = bias[col];
    v8f acc;
#pragma unroll
    for (int r = 0; r < 8; ++r) acc[r] = bv;
#pragma unroll
    for (int kk = 0; kk < 4; ++kk) {
      v16h a = frag_a(inb, HID, mt * 16, kk * 32, lane);
      v16h b = frag_b(wg, HID, nt * 16, kk * 32, lane);
      acc = __builtin_amdgcn_wmma_f32_16x16x32_f16(false, a, false, b, (short)0, acc, false, false);
    }
    int r0 = mt * 16 + ((lane & 16) ? 8 : 0);
#pragma unroll
    for (int r = 0; r < 8; ++r)
      outf[(r0 + r) * N + col] = acc[r];
  }
}

// ---------------- actor/critic heads: MLPs + log-softmax/entropy/value ----------------
__global__ __launch_bounds__(256) void head_kernel(const _Float16* __restrict__ hid,
                                                   const unsigned char* __restrict__ ws,
                                                   const int* __restrict__ action,
                                                   const float* __restrict__ ab1,
                                                   const float* __restrict__ ab2,
                                                   const float* __restrict__ ab3,
                                                   const float* __restrict__ cb1,
                                                   const float* __restrict__ cb2,
                                                   const float* __restrict__ cw3,
                                                   const float* __restrict__ cb3,
                                                   float* __restrict__ out) {
  alignas(16) __shared__ _Float16 t0b[64 * HID];
  alignas(16) __shared__ _Float16 curb[64 * HID];
  alignas(16) __shared__ _Float16 nxtb[64 * HID];
  alignas(16) __shared__ float    lgb[64 * ALOG];
  alignas(16) __shared__ float    red[512];
  const int tid = threadIdx.x, lane = tid & 31, wave = tid >> 5;
  const size_t rows0 = (size_t)blockIdx.x * 64;

  // t0 = tanh(hidden), vectorized b128 loads
  for (int i = tid * 8; i < 64 * HID; i += 256 * 8) {
    v8h x = *(const v8h*)(hid + rows0 * HID + i);
    v8h y;
#pragma unroll
    for (int k = 0; k < 8; ++k) y[k] = (_Float16)tanhf_((float)x[k]);
    *(v8h*)(t0b + i) = y;
  }
  __syncthreads();

  gemm_tanh(t0b, (const _Float16*)(ws + WS_AW1), ab1, HID, curb, lane, wave);
  __syncthreads();
  gemm_tanh(curb, (const _Float16*)(ws + WS_AW2), ab2, HID, nxtb, lane, wave);
  __syncthreads();
  gemm_lin(nxtb, (const _Float16*)(ws + WS_AW3), ab3, ALOG, lgb, lane, wave);
  __syncthreads();
  gemm_tanh(t0b, (const _Float16*)(ws + WS_CW1), cb1, HID, curb, lane, wave);
  __syncthreads();
  gemm_tanh(curb, (const _Float16*)(ws + WS_CW2), cb2, HID, nxtb, lane, wave);
  __syncthreads();

  // per (sample, group) log-softmax over 16 bins
  {
    int s = tid >> 2, g = tid & 3;
    const float* lp = &lgb[s * ALOG + g * NBINS];
    float mx = lp[0];
#pragma unroll
    for (int i = 1; i < NBINS; ++i) mx = fmaxf(mx, lp[i]);
    float se = 0.f, sxe = 0.f;
#pragma unroll
    for (int i = 0; i < NBINS; ++i) {
      float e = __expf(lp[i] - mx);
      se += e;
      sxe += lp[i] * e;
    }
    float lse = mx + __logf(se);
    int a = action[(rows0 + s) * NACT + g];
    red[tid * 2 + 0] = lp[a] - lse;      // group log-prob
    red[tid * 2 + 1] = lse - sxe / se;   // group entropy
  }
  __syncthreads();

  if (tid < 64) {
    float l = 0.f, e = 0.f;
#pragma unroll
    for (int g = 0; g < 4; ++g) {
      l += red[(tid * 4 + g) * 2 + 0];
      e += red[(tid * 4 + g) * 2 + 1];
    }
    float v = cb3[0];
    for (int k = 0; k < HID; ++k) v += cw3[k] * (float)nxtb[tid * HID + k];
    size_t o = (rows0 + tid) * 3;
    out[o + 0] = l;
    out[o + 1] = e;
    out[o + 2] = v;
  }
}

extern "C" void kernel_launch(void* const* d_in, const int* in_sizes, int n_in,
                              void* d_out, int out_size, void* d_ws, size_t ws_size,
                              hipStream_t stream) {
  (void)in_sizes; (void)n_in; (void)out_size; (void)ws_size;
  const float* obs    = (const float*)d_in[0];
  const float* done   = (const float*)d_in[1];
  const float* h0     = (const float*)d_in[2];
  const float* c0     = (const float*)d_in[3];
  const int*   action = (const int*)d_in[4];
  const float* w_ih = (const float*)d_in[5];
  const float* w_hh = (const float*)d_in[6];
  const float* b_ih = (const float*)d_in[7];
  const float* b_hh = (const float*)d_in[8];
  const float* aw1 = (const float*)d_in[9];   const float* ab1 = (const float*)d_in[10];
  const float* aw2 = (const float*)d_in[11];  const float* ab2 = (const float*)d_in[12];
  const float* aw3 = (const float*)d_in[13];  const float* ab3 = (const float*)d_in[14];
  const float* cw1 = (const float*)d_in[15];  const float* cb1 = (const float*)d_in[16];
  const float* cw2 = (const float*)d_in[17];  const float* cb2 = (const float*)d_in[18];
  const float* cw3 = (const float*)d_in[19];  const float* cb3 = (const float*)d_in[20];

  unsigned char* ws = (unsigned char*)d_ws;
  _Float16* hid = (_Float16*)(ws + WS_HID);

  const int prep_elems = G4H * KC + G4H + 3 * HID * HID + ALOG * HID + HID * HID;
  prep_kernel<<<(prep_elems + 255) / 256, 256, 0, stream>>>(w_ih, w_hh, b_ih, b_hh,
                                                            aw1, aw2, aw3, cw1, cw2, ws);
  lstm_kernel<<<BATCH / 16, 256, 0, stream>>>(obs, done, h0, c0, ws, hid);
  head_kernel<<<(T_STEPS * BATCH) / 64, 256, 0, stream>>>(hid, ws, action,
                                                          ab1, ab2, ab3, cb1, cb2, cw3, cb3,
                                                          (float*)d_out);
}
